// HeterogeneousGNN_70695161692601
// MI455X (gfx1250) — compile-verified, matrix operand order checked by reading, not run on hardware
//
#include <hip/hip_runtime.h>

// ---------------------------------------------------------------------------
// MI455X (gfx1250) fused heterogeneous-GNN forward.
// One workgroup per batch element; whole 256x128 activation resident in LDS
// across all 3 layers (HBM traffic ~= input read only, ~12us @ 23.3 TB/s, so
// the kernel is WMMA-bound: use V_WMMA_F32_16X16X32_BF16 everywhere).
// The signed linear stage is folded into ONE K=256 GEMM:
//   GELU(AggP@Wp^T - AggN@Wn^T) == GELU([AggP|AggN] @ [Wp ; -Wn]^T)
// with -Wn pre-negated in the bf16 weight prologue.
// ---------------------------------------------------------------------------

typedef __attribute__((ext_vector_type(8)))  float   v8f;
typedef __attribute__((ext_vector_type(8)))  __bf16  v8bf;
typedef __attribute__((ext_vector_type(16))) __bf16  v16bf;

#define NNODES   256
#define DIM      128
#define NLAYERS  3
#define LN_EPS   1e-5f

// LDS layouts (bf16 elements). 8-element pads skew rows by 16B across banks.
#define HT_LD    264                        // 256 + 8 (row stride, elements)
#define P_LD     264                        // 256 + 8
#define HT_BYTES (128 * HT_LD * 2)          // 67,584 B : H transposed [d][v]
#define P_BYTES  (256 * P_LD * 2)           // 135,168 B: [u][ AggP(128) | AggN(128) ]
#define SMEM_BYTES (HT_BYTES + P_BYTES)     // 202,752 B < 320 KB / WGP

__device__ __forceinline__ unsigned short f2bf(float f) {
  unsigned int u = __builtin_bit_cast(unsigned int, f);
  u += 0x7fffu + ((u >> 16) & 1u);          // round-to-nearest-even
  return (unsigned short)(u >> 16);
}
__device__ __forceinline__ float bf2f(unsigned short h) {
  unsigned int u = ((unsigned int)h) << 16;
  return __builtin_bit_cast(float, u);
}

// Build one 16x32 bf16 WMMA operand for this lane from two 16-byte loads.
// Lane half 0 (lanes 0-15) holds K = [0..7] and [16..23]; half 1 holds
// K = [8..15] and [24..31]  (CDNA5 16-bit A/B operand layout).
__device__ __forceinline__ v16bf ld32bf(const unsigned short* p) {
  v8bf lo = *(const v8bf*)p;
  v8bf hi = *(const v8bf*)(p + 16);
  return __builtin_shufflevector(lo, hi, 0,1,2,3,4,5,6,7,8,9,10,11,12,13,14,15);
}

__device__ __forceinline__ v8f wmma_bf16(v16bf a, v16bf b, v8f c) {
  return __builtin_amdgcn_wmma_f32_16x16x32_bf16(
      /*neg_a=*/false, a, /*neg_b=*/false, b,
      /*c_mod=*/(short)0, c, /*reuse_a=*/false, /*reuse_b=*/false);
}

// --------------------------- prologue kernels ------------------------------

__global__ __launch_bounds__(256)
void build_adj_kernel(const float* __restrict__ lp, const float* __restrict__ ln_,
                      unsigned short* __restrict__ Ap, unsigned short* __restrict__ An) {
  int i = blockIdx.x * 256 + threadIdx.x;            // 256*256 elements
  float ap = 1.f / (1.f + __expf(-lp[i]));
  float an = 1.f / (1.f + __expf(-ln_[i]));
  bool m = ap > an;                                   // hard mask split
  Ap[i] = f2bf(m ? ap : 0.f);
  An[i] = f2bf(m ? 0.f : an);
}

// Wcat[l][o][0:128)   =  Wp[l][o][:]
// Wcat[l][o][128:256) = -Wn[l][o][:]   (sign folded into the weights)
__global__ __launch_bounds__(256)
void build_w_kernel(const float* __restrict__ wp, const float* __restrict__ wn,
                    unsigned short* __restrict__ wcat, int n) {
  int i = blockIdx.x * 256 + threadIdx.x;            // n = 3*128*128
  if (i < n) {
    int lo = i >> 7;                                  // l*128 + o
    int d  = i & 127;
    wcat[lo * 256 + d]       = f2bf(wp[i]);
    wcat[lo * 256 + 128 + d] = f2bf(-wn[i]);
  }
}

// ------------------------------ main kernel --------------------------------

__global__ __launch_bounds__(256)
void gnn_fused_kernel(const float* __restrict__ nodef,
                      const unsigned short* __restrict__ Ap,
                      const unsigned short* __restrict__ An,
                      const unsigned short* __restrict__ Wcat,
                      const float* __restrict__ lng, const float* __restrict__ lnb,
                      const float* __restrict__ rog, const float* __restrict__ rob,
                      float* __restrict__ out) {
  extern __shared__ char smem[];
  unsigned short* Ht = (unsigned short*)smem;                  // [128][HT_LD]
  unsigned short* P  = (unsigned short*)(smem + HT_BYTES);     // [256][P_LD]

  const int tid  = threadIdx.x;
  const int lane = tid & 31;                // wave32
  const int w    = tid >> 5;                // 8 waves
  const int b    = blockIdx.x;

  const int l15  = lane & 15;
  const int koff = (lane >> 4) << 3;        // K-chunk select per lane half
  const int half = (lane >> 4) << 3;        // C/D row offset per lane half

  const v8f vzero = {0.f,0.f,0.f,0.f,0.f,0.f,0.f,0.f};

  // ---- stage 0: load H0 coalesced, convert to bf16, store transposed ----
  const float* Hb = nodef + (size_t)b * (NNODES * DIM);
  for (int e = tid * 4; e < NNODES * DIM; e += 256 * 4) {
    float4 x = *(const float4*)(Hb + e);    // 4 KB contiguous per iteration
    int v = e >> 7, d = e & 127;
    Ht[(d+0)*HT_LD + v] = f2bf(x.x);
    Ht[(d+1)*HT_LD + v] = f2bf(x.y);
    Ht[(d+2)*HT_LD + v] = f2bf(x.z);
    Ht[(d+3)*HT_LD + v] = f2bf(x.w);
  }
  __syncthreads();

  for (int l = 0; l < NLAYERS; ++l) {
    // ====== aggregation: P[:, :128] = A_pos @ H,  P[:, 128:] = A_neg @ H ===
    // wave w owns M-strips mu = 2w, 2w+1 (32 output rows), all 8 N-tiles.
    for (int mi = 0; mi < 2; ++mi) {
      const int mu = (w << 1) + mi;
      v8f accp[8], accn[8];
      #pragma unroll
      for (int n = 0; n < 8; ++n) { accp[n] = vzero; accn[n] = vzero; }

      const unsigned short* apRow = Ap + ((mu << 4) + l15) * NNODES + koff;
      const unsigned short* anRow = An + ((mu << 4) + l15) * NNODES + koff;
      for (int k0 = 0; k0 < NNODES; k0 += 32) {
        __builtin_prefetch(apRow + k0 + 32, 0, 0);   // global_prefetch_b8
        __builtin_prefetch(anRow + k0 + 32, 0, 0);
        v16bf a_p = ld32bf(apRow + k0);              // L2-resident adjacency
        v16bf a_n = ld32bf(anRow + k0);
        #pragma unroll
        for (int n = 0; n < 8; ++n) {                // reuse A over 8 N-tiles
          v16bf bb = ld32bf(&Ht[(n*16 + l15)*HT_LD + k0 + koff]);
          accp[n] = wmma_bf16(a_p, bb, accp[n]);
          accn[n] = wmma_bf16(a_n, bb, accn[n]);
        }
      }
      const int ub = (mu << 4) + half;
      #pragma unroll
      for (int n = 0; n < 8; ++n) {
        const int d = n*16 + l15;
        #pragma unroll
        for (int r = 0; r < 8; ++r) {       // C layout: VGPR r -> row r / r+8
          P[(ub + r)*P_LD + d]       = f2bf(accp[n][r]);
          P[(ub + r)*P_LD + 128 + d] = f2bf(accn[n][r]);
        }
      }
    }
    __syncthreads();

    // == linear: O = P @ Wcat^T (K=256, sign pre-folded) + GELU + LN -> Ht ==
    const unsigned short* Wl = Wcat + l * (DIM * 256);  // [o][256] row-major
    for (int mi = 0; mi < 2; ++mi) {
      const int mu = (w << 1) + mi;
      v8f acc[8];
      #pragma unroll
      for (int n = 0; n < 8; ++n) acc[n] = vzero;

      const unsigned short* pRow = P + ((mu << 4) + l15) * P_LD + koff;
      for (int k0 = 0; k0 < 256; k0 += 32) {
        v16bf a = ld32bf(pRow + k0);
        #pragma unroll
        for (int n = 0; n < 8; ++n) {
          const int o = n*16 + l15;         // B operand: Wl[o][k] K-contiguous
          acc[n] = wmma_bf16(a, ld32bf(Wl + o*256 + k0 + koff), acc[n]);
        }
      }

      // exact-erf GELU + per-row stats (rows spread across lane halves)
      float pre[8][8];
      float s[8], q[8];
      #pragma unroll
      for (int r = 0; r < 8; ++r) { s[r] = 0.f; q[r] = 0.f; }
      #pragma unroll
      for (int n = 0; n < 8; ++n) {
        #pragma unroll
        for (int r = 0; r < 8; ++r) {
          float x = acc[n][r];
          float g = 0.5f * x * (1.f + erff(x * 0.70710678118654752f));
          pre[n][r] = g; s[r] += g; q[r] += g * g;
        }
      }
      #pragma unroll
      for (int r = 0; r < 8; ++r) {
        float ss = s[r], qq = q[r];
        #pragma unroll
        for (int m = 1; m < 16; m <<= 1) {  // reduce within 16-lane half
          ss += __shfl_xor(ss, m);
          qq += __shfl_xor(qq, m);
        }
        float mean = ss * (1.f / 128.f);
        float var  = qq * (1.f / 128.f) - mean * mean;
        float rs   = rsqrtf(var + LN_EPS);
        const int u = (mu << 4) + r + half;
        #pragma unroll
        for (int n = 0; n < 8; ++n) {
          const int d = n*16 + l15;
          float y = (pre[n][r] - mean) * rs * lng[l*DIM + d] + lnb[l*DIM + d];
          Ht[d*HT_LD + u] = f2bf(y);        // transposed store for next layer
        }
      }
    }
    __syncthreads();
  }

  // ---- readout: mean over nodes, then LayerNorm over DIM ----
  float* hG = (float*)P;                    // reuse agg buffer
  if (tid < DIM) {
    float sum = 0.f;
    for (int v = 0; v < NNODES; ++v) sum += bf2f(Ht[tid*HT_LD + v]);
    hG[tid] = sum * (1.f / 256.f);
  }
  __syncthreads();
  if (tid < DIM) {
    float mean = 0.f, m2 = 0.f;
    for (int d2 = 0; d2 < DIM; ++d2) { float x = hG[d2]; mean += x; m2 += x * x; }
    mean *= (1.f / 128.f);
    float var = m2 * (1.f / 128.f) - mean * mean;
    float rs  = rsqrtf(var + LN_EPS);
    out[(size_t)b * DIM + tid] = (hG[tid] - mean) * rs * rog[tid] + rob[tid];
  }
}

// ------------------------------ host launch --------------------------------

extern "C" void kernel_launch(void* const* d_in, const int* in_sizes, int n_in,
                              void* d_out, int out_size, void* d_ws, size_t ws_size,
                              hipStream_t stream) {
  const float* nodef = (const float*)d_in[0];
  const float* alp   = (const float*)d_in[1];
  const float* aln   = (const float*)d_in[2];
  const float* wp    = (const float*)d_in[3];
  const float* wn    = (const float*)d_in[4];
  const float* lng   = (const float*)d_in[5];
  const float* lnb   = (const float*)d_in[6];
  const float* rog   = (const float*)d_in[7];
  const float* rob   = (const float*)d_in[8];
  float* out = (float*)d_out;

  const int B = in_sizes[0] / (NNODES * DIM);

  // Workspace layout (bf16): A_pos | A_neg | Wcat  (~450 KB, L2-resident)
  unsigned short* Ap   = (unsigned short*)d_ws;
  unsigned short* An   = Ap + NNODES * NNODES;
  unsigned short* Wcat = An + NNODES * NNODES;       // 3*128*256 bf16

  build_adj_kernel<<<(NNODES * NNODES) / 256, 256, 0, stream>>>(alp, aln, Ap, An);
  const int nw = NLAYERS * DIM * DIM;
  build_w_kernel<<<(nw + 255) / 256, 256, 0, stream>>>(wp, wn, Wcat, nw);

  hipFuncSetAttribute((const void*)gnn_fused_kernel,
                      hipFuncAttributeMaxDynamicSharedMemorySize, SMEM_BYTES);
  gnn_fused_kernel<<<B, 256, SMEM_BYTES, stream>>>(
      nodef, Ap, An, Wcat, lng, lnb, rog, rob, out);
}
